// AttentionTSP_44384192037555
// MI455X (gfx1250) — compile-verified, matrix operand order checked by reading, not run on hardware
//
#include <hip/hip_runtime.h>

// ---------------------------------------------------------------------------
// Types / helpers for CDNA5 WMMA (gfx1250, wave32)
// ---------------------------------------------------------------------------
typedef __attribute__((ext_vector_type(16))) __bf16 v16bf;
typedef __attribute__((ext_vector_type(8)))  float  v8f;

union FragAB { v16bf v; unsigned u[8]; };
union FragC  { v8f   v; float    f[8]; };

__device__ __forceinline__ unsigned short f2bf(float x) {
    unsigned u = __float_as_uint(x);
    u += 0x7FFFu + ((u >> 16) & 1u);     // round-to-nearest-even
    return (unsigned short)(u >> 16);
}
__device__ __forceinline__ unsigned packbf(float a, float b) {
    return (unsigned)f2bf(a) | ((unsigned)f2bf(b) << 16);
}
// bf16 A/B fragment: VGPR v at lane (lm, hi) holds K-pair index:
// v0..3 -> pairs {0..3}+4*hi ; v4..7 -> pairs {8..11}+4*hi
__device__ __forceinline__ int pairIdx(int v, int hi) {
    return (v & 3) + ((v >> 2) << 3) + (hi << 2);
}
__device__ __forceinline__ v8f wmma_bf16(v16bf a, v16bf b, v8f c) {
    return __builtin_amdgcn_wmma_f32_16x16x32_bf16(false, a, false, b, (short)0, c, false, false);
}

// ---- CDNA5 async global->LDS DMA (ASYNCcnt path) --------------------------
// LDS address = low 32 bits of the generic pointer (ISA 10.2).
__device__ __forceinline__ void async_copy16(void* lds_dst, const void* gsrc) {
    unsigned l = (unsigned)(size_t)lds_dst;
    unsigned long long g = (unsigned long long)(size_t)gsrc;
    asm volatile("global_load_async_to_lds_b128 %0, %1, off"
                 :: "v"(l), "v"(g) : "memory");
}
__device__ __forceinline__ void wait_async0() {
#if __has_builtin(__builtin_amdgcn_s_wait_asynccnt)
    __builtin_amdgcn_s_wait_asynccnt(0);
#else
    asm volatile("s_wait_asynccnt 0" ::: "memory");
#endif
}

#define NEGV  (-1e9f)
#define SCG   0.17677669529663689f   // 1/sqrt(32)
#define SCP   0.08838834764831845f   // 1/sqrt(128)

// ---------------------------------------------------------------------------
// Graph embedding: h[t,e] = x0*emb[0,e] + x1*emb[1,e]
// ---------------------------------------------------------------------------
__global__ __launch_bounds__(256) void embed_kernel(const float* __restrict__ x,
                                                    const float* __restrict__ emb,
                                                    float* __restrict__ h) {
    int gid = blockIdx.x * 256 + threadIdx.x;        // BS*32 threads
    int t = gid >> 5, eq = (gid & 31) * 4;
    float x0 = x[(size_t)t * 2 + 0];
    float x1 = x[(size_t)t * 2 + 1];
    float4 e0 = *(const float4*)(emb + eq);
    float4 e1 = *(const float4*)(emb + 128 + eq);
    float4 r;
    r.x = x0 * e0.x + x1 * e1.x;
    r.y = x0 * e0.y + x1 * e1.y;
    r.z = x0 * e0.z + x1 * e1.z;
    r.w = x0 * e0.w + x1 * e1.w;
    *(float4*)(h + (size_t)t * 128 + eq) = r;
}

// ---------------------------------------------------------------------------
// Generic GEMM: C[M,N] = act(A[M,K=128] @ W[N,128]^T + bias (+Res))
// 128x128 block tile, 8 waves, wave tile 32x64 (2x4 WMMA 16x16 tiles).
// Pipeline per K-chunk(32):
//   async-DMA f32 tiles -> LDS (ASYNCcnt), repack once to bf16 pairs,
//   issue DMA for chunk c+1, then WMMA on chunk c (DMA || WMMA overlap
//   with single buffers thanks to the decoupled async engine).
// ---------------------------------------------------------------------------
__device__ __forceinline__ void issue_chunk(const float* A, int lda, const float* W,
                                            int m0, int n0, int c,
                                            float* Af, float* Wf) {
    for (int i = threadIdx.x; i < 2048; i += 256) {
        int isW = i >> 10, j = i & 1023;
        int r = j >> 3, q = j & 7;
        const float* src = (isW ? W + (size_t)(n0 + r) * 128
                                : A + (size_t)(m0 + r) * lda) + c * 32 + q * 4;
        float* dst = (isW ? Wf : Af) + r * 32 + q * 4;
        async_copy16(dst, src);
        if (c < 3) __builtin_prefetch(src + 32, 0, 0);   // next-chunk hint
    }
}

__global__ __launch_bounds__(256) void gemm_f32_bf16(
    const float* __restrict__ A, int lda,
    const float* __restrict__ W,            // [N,128] row-major
    const float* __restrict__ bias,
    const float* __restrict__ Res,          // may be null; ld = ldc
    float* __restrict__ C, int ldc,
    int relu) {
    __shared__ float    Af[128 * 32];       // async DMA dst (f32 chunk)
    __shared__ float    Wf[128 * 32];
    __shared__ unsigned Ap[128 * 17];       // bf16 pairs, padded stride
    __shared__ unsigned Wp[128 * 17];
    const int tid = threadIdx.x;
    const int lane = tid & 31, wave = tid >> 5;
    const int lm = lane & 15, hi = lane >> 4;
    const int m0 = blockIdx.x * 128;
    const int n0 = blockIdx.y * 128;
    const int wm = (wave >> 1) * 32;        // 0,32,64,96
    const int wn = (wave & 1) * 64;         // 0,64

    v8f acc[2][4];
#pragma unroll
    for (int i = 0; i < 2; ++i)
#pragma unroll
        for (int j = 0; j < 4; ++j) acc[i][j] = (v8f){0, 0, 0, 0, 0, 0, 0, 0};

    issue_chunk(A, lda, W, m0, n0, 0, Af, Wf);          // prologue DMA

    for (int c = 0; c < 4; ++c) {           // K chunks of 32
        wait_async0();                      // own chunk-c copies landed
        __syncthreads();                    // all waves' copies landed; bf16 bufs free
        // repack once: f32 LDS -> bf16-pair LDS (1 pack per element)
        for (int i = tid; i < 4096; i += 256) {
            int isW = i >> 11, j = i & 2047;
            int r = j >> 4, p = j & 15;
            const float* src = (isW ? Wf : Af) + r * 32 + 2 * p;
            (isW ? Wp : Ap)[r * 17 + p] = packbf(src[0], src[1]);
        }
        __syncthreads();                    // repack done: f32 bufs reusable
        if (c < 3) issue_chunk(A, lda, W, m0, n0, c + 1, Af, Wf);  // DMA || WMMA

        FragAB afr[2], bfr[4];
#pragma unroll
        for (int mt = 0; mt < 2; ++mt)
#pragma unroll
            for (int v = 0; v < 8; ++v)
                afr[mt].u[v] = Ap[(wm + mt * 16 + lm) * 17 + pairIdx(v, hi)];
#pragma unroll
        for (int nt = 0; nt < 4; ++nt)
#pragma unroll
            for (int v = 0; v < 8; ++v)
                bfr[nt].u[v] = Wp[(wn + nt * 16 + lm) * 17 + pairIdx(v, hi)];
#pragma unroll
        for (int mt = 0; mt < 2; ++mt)
#pragma unroll
            for (int nt = 0; nt < 4; ++nt)
                acc[mt][nt] = wmma_bf16(afr[mt].v, bfr[nt].v, acc[mt][nt]);
        __syncthreads();                    // bf16 reads done before next repack
    }
    // epilogue: C layout — lane lm = col, reg j = row (+8 for hi half)
#pragma unroll
    for (int mt = 0; mt < 2; ++mt)
#pragma unroll
        for (int nt = 0; nt < 4; ++nt) {
            FragC fc; fc.v = acc[mt][nt];
            int n = n0 + wn + nt * 16 + lm;
            float bv = bias ? bias[n] : 0.f;
#pragma unroll
            for (int j = 0; j < 8; ++j) {
                int m = m0 + wm + mt * 16 + j + hi * 8;
                float r = fc.f[j] + bv;
                if (Res) r += Res[(size_t)m * ldc + n];
                if (relu) r = fmaxf(r, 0.f);
                C[(size_t)m * ldc + n] = r;
            }
        }
}

// ---------------------------------------------------------------------------
// Encoder self-attention, one block per (batch, head). S=128, DH=32.
// scores via WMMA (K=32, one chunk); softmax in accumulator registers via
// shfl_xor (a wave owns complete rows); P@V via WMMA. O overwrites the
// q-slice of qkv in place (each block owns its slice).
// ---------------------------------------------------------------------------
__global__ __launch_bounds__(256) void attn_kernel(float* __restrict__ qkv) {
    __shared__ unsigned Qs[128 * 17];
    __shared__ unsigned Ks[128 * 17];
    __shared__ unsigned Vt[32 * 65];            // V transposed: [d][s-pairs]
    __shared__ unsigned short Ps[128 * 130];    // probs bf16

    const int tid = threadIdx.x, lane = tid & 31, wave = tid >> 5;
    const int lm = lane & 15, hi = lane >> 4;
    const int b = blockIdx.x >> 2, hd = blockIdx.x & 3;
    float* base = qkv + (size_t)b * 128 * 384;
    const float* qb = base + hd * 32;
    const float* kb = base + 128 + hd * 32;
    const float* vb = base + 256 + hd * 32;

    for (int i = tid; i < 128 * 16; i += 256) {
        int r = i >> 4, p = i & 15;
        const float* pq = qb + (size_t)r * 384 + 2 * p;
        Qs[r * 17 + p] = packbf(pq[0], pq[1]);
        const float* pk = kb + (size_t)r * 384 + 2 * p;
        Ks[r * 17 + p] = packbf(pk[0], pk[1]);
    }
    for (int i = tid; i < 32 * 64; i += 256) {
        int d = i >> 6, p = i & 63;
        Vt[d * 65 + p] = packbf(vb[(size_t)(2 * p) * 384 + d],
                                vb[(size_t)(2 * p + 1) * 384 + d]);
    }
    __syncthreads();

    const int mw = wave * 16;                    // this wave's 16 query rows
    FragAB afr;
#pragma unroll
    for (int v = 0; v < 8; ++v) afr.u[v] = Qs[(mw + lm) * 17 + pairIdx(v, hi)];
    FragC sc[8];
#pragma unroll
    for (int nt = 0; nt < 8; ++nt) {
        FragAB bfr;
#pragma unroll
        for (int v = 0; v < 8; ++v) bfr.u[v] = Ks[(nt * 16 + lm) * 17 + pairIdx(v, hi)];
        v8f z = (v8f){0, 0, 0, 0, 0, 0, 0, 0};
        sc[nt].v = wmma_bf16(afr.v, bfr.v, z);
    }
    // register softmax: row (mw + j + 8*hi) lives across 16 lanes x 8 regs
#pragma unroll
    for (int j = 0; j < 8; ++j) {
        float mx = -1e30f;
#pragma unroll
        for (int nt = 0; nt < 8; ++nt) mx = fmaxf(mx, sc[nt].f[j]);
        for (int d = 1; d < 16; d <<= 1) mx = fmaxf(mx, __shfl_xor(mx, d, 32));
        float e[8], sum = 0.f;
#pragma unroll
        for (int nt = 0; nt < 8; ++nt) { e[nt] = __expf((sc[nt].f[j] - mx) * SCG); sum += e[nt]; }
        for (int d = 1; d < 16; d <<= 1) sum += __shfl_xor(sum, d, 32);
        float inv = 1.f / sum;
        int m = mw + j + hi * 8;
#pragma unroll
        for (int nt = 0; nt < 8; ++nt) Ps[m * 130 + nt * 16 + lm] = f2bf(e[nt] * inv);
    }
    __syncthreads();
    // O = P @ V  (M=128, N=32, K=128)
#pragma unroll
    for (int nt = 0; nt < 2; ++nt) {
        v8f o = (v8f){0, 0, 0, 0, 0, 0, 0, 0};
#pragma unroll
        for (int c = 0; c < 4; ++c) {
            FragAB pa, vv;
#pragma unroll
            for (int v = 0; v < 8; ++v) {
                pa.u[v] = *(const unsigned*)&Ps[(mw + lm) * 130 + 2 * (c * 16 + pairIdx(v, hi))];
                vv.u[v] = Vt[(nt * 16 + lm) * 65 + c * 16 + pairIdx(v, hi)];
            }
            o = wmma_bf16(pa.v, vv.v, o);
        }
        FragC fo; fo.v = o;
#pragma unroll
        for (int j = 0; j < 8; ++j) {
            int s = mw + j + hi * 8;
            base[(size_t)s * 384 + hd * 32 + nt * 16 + lm] = fo.f[j];
        }
    }
}

// ---------------------------------------------------------------------------
// Deterministic BatchNorm (training mode) over [262144, 128]
// ---------------------------------------------------------------------------
__global__ __launch_bounds__(256) void bn_stats(const float* __restrict__ X,
                                                float* __restrict__ part) {
    int c = threadIdx.x & 127, hh = threadIdx.x >> 7;
    size_t r0 = (size_t)blockIdx.x * 2048 + (size_t)hh * 1024;
    float s = 0.f, q = 0.f;
    for (int r = 0; r < 1024; ++r) {
        float x = X[(r0 + r) * 128 + c];
        s += x; q += x * x;
    }
    part[blockIdx.x * 256 + threadIdx.x] = s;
    part[32768 + blockIdx.x * 256 + threadIdx.x] = q;
}
__global__ __launch_bounds__(128) void bn_finalize(const float* __restrict__ part,
                                                   float* __restrict__ st) {
    int c = threadIdx.x;
    float s = 0.f, q = 0.f;
    for (int b = 0; b < 128; ++b) {
        s += part[b * 256 + c] + part[b * 256 + 128 + c];
        q += part[32768 + b * 256 + c] + part[32768 + b * 256 + 128 + c];
    }
    float inv = 1.f / 262144.f;
    float m = s * inv, v = q * inv - m * m;
    st[c] = m;
    st[128 + c] = rsqrtf(v + 1e-5f);
}
__global__ __launch_bounds__(256) void bn_apply(float* __restrict__ X,
                                                const float* __restrict__ st,
                                                const float* __restrict__ g,
                                                const float* __restrict__ bt) {
    size_t idx = (size_t)blockIdx.x * 256 + threadIdx.x;
    int c = (int)(idx & 127);
    X[idx] = (X[idx] - st[c]) * st[128 + c] * g[c] + bt[c];
}

// h row-sum over S (for left0)
__global__ __launch_bounds__(128) void rowsum_kernel(const float* __restrict__ h,
                                                     float* __restrict__ hs) {
    int b = blockIdx.x, e = threadIdx.x;
    const float* p = h + (size_t)b * 128 * 128 + e;
    float s = 0.f;
    for (int i = 0; i < 128; ++i) s += p[(size_t)i * 128];
    hs[(size_t)b * 128 + e] = s;
}

// ---------------------------------------------------------------------------
// Decoder: 16 batch rows per block so every linear is a [16,128]x[128,128]
// WMMA GEMM (8 waves x 1 N-tile x 4 K-chunks). 128 sequential steps in-kernel.
// ---------------------------------------------------------------------------
struct DecP {
    const float *h, *Kg, *Vg, *Kp, *hsum;
    const float *gqw, *gqb, *gow, *gob;
    const float *pqw, *pqb;
    const float *vww, *vwb, *hqw, *hqb, *mw, *mb, *cw, *cb;
    const float *h1w, *h1b, *h2w, *h2b, *initw;
    float* out;
};

// stage [16,128] f32 LDS -> bf16-pair LDS (optionally tanh)
__device__ __forceinline__ void stage16(const float* src, unsigned* Xs, int do_tanh) {
    for (int i = threadIdx.x; i < 16 * 64; i += 256) {
        int m = i >> 6, p = i & 63;
        float a = src[m * 128 + 2 * p], b = src[m * 128 + 2 * p + 1];
        if (do_tanh) { a = tanhf(a); b = tanhf(b); }
        Xs[m * 65 + p] = packbf(a, b);
    }
}
// Y[16,128] (mode: 0 set, 1 add, 2 sub) = staged-X @ W^T + bias*bscale
__device__ __forceinline__ void gemm16(const unsigned* Xs, const float* __restrict__ W,
                                       const float* __restrict__ bias, float bscale,
                                       float* Y, int mode) {
    int lane = threadIdx.x & 31, wave = threadIdx.x >> 5;
    int lm = lane & 15, hi = lane >> 4;
    int n = wave * 16 + lm;
    v8f acc = (v8f){0, 0, 0, 0, 0, 0, 0, 0};
#pragma unroll
    for (int c = 0; c < 4; ++c) {
        FragAB a, b;
        const float* wr = W + (size_t)n * 128 + c * 32;
#pragma unroll
        for (int v = 0; v < 8; ++v) {
            int p = pairIdx(v, hi);
            a.u[v] = Xs[lm * 65 + c * 16 + p];
            float2 w2 = *(const float2*)(wr + 2 * p);
            b.u[v] = packbf(w2.x, w2.y);
        }
        acc = wmma_bf16(a.v, b.v, acc);
    }
    FragC fc; fc.v = acc;
    float bv = bias[n] * bscale;
#pragma unroll
    for (int j = 0; j < 8; ++j) {
        int m = j + hi * 8;
        float r = fc.f[j] + bv;
        float* y = &Y[m * 128 + n];
        if (mode == 0) *y = r;
        else if (mode == 1) *y += r;
        else *y -= r;
    }
}

__global__ __launch_bounds__(256) void decoder_kernel(DecP P) {
    __shared__ float query[16 * 128];
    __shared__ float cvec[16 * 128];
    __shared__ float lvec[16 * 128];
    __shared__ float pool[8192];          // transient: bufA | bufB | big(16KB)
    __shared__ unsigned Xs[16 * 65];
    __shared__ unsigned maskb[16 * 4];    // 128-bit mask per row
    __shared__ int chosen[16];
    __shared__ float vw0[128];

    const int tid = threadIdx.x;
    const int b0 = blockIdx.x * 16;
    float* bufA = pool;
    float* bufB = pool + 2048;
    float* big  = pool + 4096;            // 4096 floats

    // ------------- init -------------
    for (int i = tid; i < 64; i += 256) maskb[i] = 0u;
    for (int i = tid; i < 2048; i += 256) cvec[i] = 0.f;
    if (tid < 128) {
        float s = P.vwb[tid];
        for (int k = 0; k < 128; ++k) s += P.vww[tid * 128 + k] * P.initw[k];
        vw0[tid] = s;
    }
    for (int i = tid; i < 2048; i += 256) {
        int r = i >> 7, e = i & 127;
        bufA[i] = P.hsum[(size_t)(b0 + r) * 128 + e];
    }
    __syncthreads();
    stage16(bufA, Xs, 0);                       // hsum
    __syncthreads();
    gemm16(Xs, P.mw, P.mb, 128.f, lvec, 0);     // left0 = Wm@hsum + S*b
    __syncthreads();
    stage16(lvec, Xs, 1);                       // tanh(left0)
    __syncthreads();
    gemm16(Xs, P.h2w, P.h2b, 1.f, bufB, 0);
    __syncthreads();
    for (int i = tid; i < 2048; i += 256) {     // + b_h1 (tanh(0) path) + vw0
        int e = i & 127;
        bufB[i] += P.h1b[e] + vw0[e];
    }
    __syncthreads();
    stage16(bufB, Xs, 0);
    __syncthreads();
    gemm16(Xs, P.hqw, P.hqb, 1.f, query, 0);    // q0
    __syncthreads();

    // ------------- 128 decode steps -------------
    for (int step = 0; step < 128; ++step) {
        stage16(query, Xs, 0);
        __syncthreads();
        gemm16(Xs, P.gqw, P.gqb, 1.f, bufA, 0); // qg
        __syncthreads();

        // glimpse attention, two heads at a time (big = [16][2][128])
        for (int hp = 0; hp < 2; ++hp) {
            for (int i = tid; i < 2048; i += 256) {
                int r = i >> 7, s = i & 127;
                int msk = (maskb[r * 4 + (s >> 5)] >> (s & 31)) & 1;
                const float* kr = P.Kg + ((size_t)(b0 + r) * 128 + s) * 128 + hp * 64;
                const float* q = bufA + r * 128 + hp * 64;
                float l0 = 0.f, l1 = 0.f;
                for (int d = 0; d < 32; ++d) { l0 += q[d] * kr[d]; l1 += q[32 + d] * kr[32 + d]; }
                big[(r * 2 + 0) * 128 + s] = msk ? NEGV : l0 * SCG;
                big[(r * 2 + 1) * 128 + s] = msk ? NEGV : l1 * SCG;
            }
            __syncthreads();
            if (tid < 32) {
                float* row = big + tid * 128;
                float mx = row[0];
                for (int s = 1; s < 128; ++s) mx = fmaxf(mx, row[s]);
                float sum = 0.f;
                for (int s = 0; s < 128; ++s) { float e = __expf(row[s] - mx); row[s] = e; sum += e; }
                float inv = 1.f / sum;
                for (int s = 0; s < 128; ++s) row[s] *= inv;
            }
            __syncthreads();
            for (int i = tid; i < 1024; i += 256) {     // ret slice
                int r = i >> 6, d = i & 63;
                const float* a = big + (r * 2 + (d >> 5)) * 128;
                const float* vv = P.Vg + ((size_t)(b0 + r) * 128) * 128 + hp * 64 + d;
                float s = 0.f;
                for (int t = 0; t < 128; ++t) s += a[t] * vv[(size_t)t * 128];
                bufB[r * 128 + hp * 64 + d] = s;
            }
            __syncthreads();
        }
        stage16(bufB, Xs, 0);                    // ret
        __syncthreads();
        gemm16(Xs, P.gow, P.gob, 1.f, bufA, 0);  // nq
        __syncthreads();
        stage16(bufA, Xs, 0);
        __syncthreads();
        gemm16(Xs, P.pqw, P.pqb, 1.f, bufB, 0);  // qp
        __syncthreads();

        // pointer logits + argmax decode
        float* plog = big;
        for (int i = tid; i < 2048; i += 256) {
            int r = i >> 7, s = i & 127;
            int msk = (maskb[r * 4 + (s >> 5)] >> (s & 31)) & 1;
            const float* kr = P.Kp + ((size_t)(b0 + r) * 128 + s) * 128;
            const float* q = bufB + r * 128;
            float l = 0.f;
            for (int d = 0; d < 128; ++d) l += q[d] * kr[d];
            plog[i] = msk ? NEGV : 10.f * tanhf(l * SCP);
        }
        __syncthreads();
        if (tid < 16) {
            const float* row = plog + tid * 128;
            float mx = row[0]; int arg = 0;
            for (int s = 1; s < 128; ++s) if (row[s] > mx) { mx = row[s]; arg = s; }
            float sum = 0.f;
            for (int s = 0; s < 128; ++s) sum += __expf(row[s] - mx);
            chosen[tid] = arg;
            maskb[tid * 4 + (arg >> 5)] |= (1u << (arg & 31));
            P.out[(size_t)(b0 + tid) * 128 + step] = -__logf(sum);          // logp
            P.out[(size_t)2048 * 128 + (size_t)(b0 + tid) * 128 + step] = (float)arg;
        }
        __syncthreads();
        // gather chosen embeddings
        for (int i = tid; i < 2048; i += 256) {
            int r = i >> 7, e = i & 127;
            bufA[i] = P.h[((size_t)(b0 + r) * 128 + chosen[r]) * 128 + e];
        }
        __syncthreads();
        stage16(bufA, Xs, 0);                    // ch
        __syncthreads();
        gemm16(Xs, P.cw, P.cb, 1.f, cvec, 1);    // chosen_vec += Wc@ch+b
        gemm16(Xs, P.mw, P.mb, 1.f, lvec, 2);    // left_vec  -= Wm@ch+b
        gemm16(Xs, P.vww, P.vwb, 1.f, bufB, 0);  // tmp = Wv@ch+b
        __syncthreads();
        stage16(cvec, Xs, 1);
        __syncthreads();
        gemm16(Xs, P.h1w, P.h1b, 1.f, bufB, 1);  // += Wh1@tanh(cv)+b
        __syncthreads();
        stage16(lvec, Xs, 1);
        __syncthreads();
        gemm16(Xs, P.h2w, P.h2b, 1.f, bufB, 1);  // += Wh2@tanh(lv)+b
        __syncthreads();
        stage16(bufB, Xs, 0);
        __syncthreads();
        gemm16(Xs, P.hqw, P.hqb, 1.f, query, 0); // query = Whq@tmp+b
        __syncthreads();
    }
}

// ---------------------------------------------------------------------------
// Launch
// ---------------------------------------------------------------------------
extern "C" void kernel_launch(void* const* d_in, const int* in_sizes, int n_in,
                              void* d_out, int out_size, void* d_ws, size_t ws_size,
                              hipStream_t stream) {
    (void)in_sizes; (void)n_in; (void)out_size; (void)ws_size;
    const float* X   = (const float*)d_in[0];
    const float* emb = (const float*)d_in[1];

    float* ws = (float*)d_ws;
    const size_t BSE = (size_t)2048 * 128 * 128;
    float* H    = ws;                 // [BS,128]
    float* QKV  = ws + BSE;           // [BS,384] (later Kg|Vg|Kp)
    float* T1   = ws + 4 * BSE;       // [BS,128]
    float* HSUM = ws + 5 * BSE;       // [B,128]
    float* PART = HSUM + (size_t)2048 * 128;   // 65536
    float* STAT = PART + 65536;                // 256

    embed_kernel<<<32768, 256, 0, stream>>>(X, emb, H);

    for (int L = 0; L < 2; ++L) {
        const float* inw  = (const float*)d_in[2 + L * 10 + 0];
        const float* inb  = (const float*)d_in[2 + L * 10 + 1];
        const float* outw = (const float*)d_in[2 + L * 10 + 2];
        const float* outb = (const float*)d_in[2 + L * 10 + 3];
        const float* f1w  = (const float*)d_in[2 + L * 10 + 4];
        const float* f1b  = (const float*)d_in[2 + L * 10 + 5];
        const float* f2w  = (const float*)d_in[2 + L * 10 + 6];
        const float* f2b  = (const float*)d_in[2 + L * 10 + 7];
        const float* gam  = (const float*)d_in[2 + L * 10 + 8];
        const float* bet  = (const float*)d_in[2 + L * 10 + 9];

        gemm_f32_bf16<<<dim3(2048, 3), 256, 0, stream>>>(H, 128, inw, inb, nullptr, QKV, 384, 0);
        attn_kernel<<<8192, 256, 0, stream>>>(QKV);                 // O -> q-slice of QKV
        gemm_f32_bf16<<<dim3(2048, 1), 256, 0, stream>>>(QKV, 384, outw, outb, H, T1, 128, 0);
        gemm_f32_bf16<<<dim3(2048, 1), 256, 0, stream>>>(T1, 128, f1w, f1b, nullptr, QKV, 128, 1);
        gemm_f32_bf16<<<dim3(2048, 1), 256, 0, stream>>>(QKV, 128, f2w, f2b, T1, H, 128, 0);
        bn_stats<<<128, 256, 0, stream>>>(H, PART);
        bn_finalize<<<1, 128, 0, stream>>>(PART, STAT);
        bn_apply<<<131072, 256, 0, stream>>>(H, STAT, gam, bet);
    }

    // decoder precompute: glimpse K/V, pointer K, h row-sums
    gemm_f32_bf16<<<dim3(2048, 1), 256, 0, stream>>>(H, 128, (const float*)d_in[25], (const float*)d_in[26], nullptr, QKV, 128, 0);
    gemm_f32_bf16<<<dim3(2048, 1), 256, 0, stream>>>(H, 128, (const float*)d_in[27], (const float*)d_in[28], nullptr, QKV + BSE, 128, 0);
    gemm_f32_bf16<<<dim3(2048, 1), 256, 0, stream>>>(H, 128, (const float*)d_in[33], (const float*)d_in[34], nullptr, QKV + 2 * BSE, 128, 0);
    rowsum_kernel<<<2048, 128, 0, stream>>>(H, HSUM);

    DecP P;
    P.h = H; P.Kg = QKV; P.Vg = QKV + BSE; P.Kp = QKV + 2 * BSE; P.hsum = HSUM;
    P.gqw = (const float*)d_in[23]; P.gqb = (const float*)d_in[24];
    P.gow = (const float*)d_in[29]; P.gob = (const float*)d_in[30];
    P.pqw = (const float*)d_in[31]; P.pqb = (const float*)d_in[32];
    P.vww = (const float*)d_in[35]; P.vwb = (const float*)d_in[36];
    P.hqw = (const float*)d_in[37]; P.hqb = (const float*)d_in[38];
    P.mw  = (const float*)d_in[39]; P.mb  = (const float*)d_in[40];
    P.cw  = (const float*)d_in[41]; P.cb  = (const float*)d_in[42];
    P.h1w = (const float*)d_in[43]; P.h1b = (const float*)d_in[44];
    P.h2w = (const float*)d_in[45]; P.h2b = (const float*)d_in[46];
    P.initw = (const float*)d_in[22];
    P.out = (float*)d_out;
    decoder_kernel<<<128, 256, 0, stream>>>(P);
}
